// TransformerEncoder_59021440582093
// MI455X (gfx1250) — matrix operand order repaired and checked by reading, not compiled, not run on hardware
//
#include <hip/hip_runtime.h>

// ---------------- problem constants ----------------
#define BB   256
#define LL   256
#define DD   512
#define HH   8
#define NLAYER 6
#define FFD  2048
#define DHD  64
#define MM   (BB*LL)      // 65536 rows
#define NATOMS 128
#define NTOT (BB*NATOMS)  // 32768 atoms

// ---------------- types ----------------
typedef __bf16 bf16;
typedef __attribute__((ext_vector_type(16))) __bf16 v16bf;
typedef __attribute__((ext_vector_type(8)))  float  v8f;
typedef __attribute__((ext_vector_type(4)))  unsigned uv4;

__device__ inline bf16 f2bf(float f) {
  unsigned u = __builtin_bit_cast(unsigned, f);
  unsigned r = (u + 0x7FFFu + ((u >> 16) & 1u)) >> 16;
  return __builtin_bit_cast(bf16, (unsigned short)r);
}

// ---------------- weight convert + transpose: dst[l][n][k] = (bf16)src[l][k][n] ----------------
__global__ __launch_bounds__(256) void wconv_kernel(const float* __restrict__ src,
                                                    bf16* __restrict__ dst,
                                                    int K, int N, int total) {
  int i = blockIdx.x * 256 + threadIdx.x;
  if (i >= total) return;
  int kn = K * N;
  int l = i / kn, rem = i - l * kn;
  int k = rem / N, n = rem - k * N;
  dst[(size_t)l * kn + (size_t)n * K + k] = f2bf(src[i]);
}

// ---------------- embedding + key-padding mask ----------------
__global__ __launch_bounds__(256) void embed_kernel(const int* __restrict__ x,
                                                    const float* __restrict__ emb,
                                                    const float* __restrict__ rate1,
                                                    float* __restrict__ h,
                                                    float* __restrict__ mask) {
  int m = blockIdx.x;                 // m = b*LL + l
  int b = m >> 8, l = m & 255;
  int tok = x[l * BB + b];            // x is [L,B]
  float r1 = rate1[0];
  const float* e = emb + (size_t)tok * DD;
  float* hr = h + (size_t)m * DD;
  for (int d = threadIdx.x; d < DD; d += 256) hr[d] = r1 * e[d];
  if (threadIdx.x == 0) mask[m] = (tok == 0) ? -1e9f : 0.0f;
}

// ---------------- atom feature scatter-add (unique (b,idx) pairs) ----------------
__global__ __launch_bounds__(256) void atom_kernel(const float* __restrict__ coord,
                                                   const int* __restrict__ tok,
                                                   const int* __restrict__ aidx,
                                                   const int* __restrict__ bidx,
                                                   const float* __restrict__ emb,
                                                   const float* __restrict__ cw,
                                                   const float* __restrict__ cb,
                                                   const float* __restrict__ rate2,
                                                   float* __restrict__ h) {
  int n = blockIdx.x;
  float c0 = coord[n * 3 + 0], c1 = coord[n * 3 + 1], c2 = coord[n * 3 + 2];
  int t = tok[n];
  int row = bidx[n] * LL + aidx[n] + 1;
  float r2 = rate2[0];
  const float* e = emb + (size_t)t * DD;
  float* hr = h + (size_t)row * DD;
  for (int d = threadIdx.x; d < DD; d += 256) {
    float v = e[d] + c0 * cw[d] + c1 * cw[DD + d] + c2 * cw[2 * DD + d] + cb[d];
    hr[d] += r2 * v;
  }
}

// ---------------- layernorm (wave per row) -> bf16 ----------------
__global__ __launch_bounds__(256) void ln_kernel(const float* __restrict__ X,
                                                 const float* __restrict__ g,
                                                 const float* __restrict__ bta,
                                                 bf16* __restrict__ O) {
  int w = threadIdx.x >> 5, lane = threadIdx.x & 31;
  int row = blockIdx.x * 8 + w;
  const float* x = X + (size_t)row * DD;
  float s = 0.f, ss = 0.f, vals[16];
#pragma unroll
  for (int j = 0; j < 16; ++j) { float v = x[lane + j * 32]; vals[j] = v; s += v; ss += v * v; }
#pragma unroll
  for (int o = 16; o >= 1; o >>= 1) { s += __shfl_xor(s, o); ss += __shfl_xor(ss, o); }
  float mean = s * (1.f / DD);
  float var = ss * (1.f / DD) - mean * mean;
  float rstd = rsqrtf(var + 1e-6f);
  bf16* o = O + (size_t)row * DD;
#pragma unroll
  for (int j = 0; j < 16; ++j) {
    int d = lane + j * 32;
    o[d] = f2bf((vals[j] - mean) * rstd * g[d] + bta[d]);
  }
}

// ---------------- final layernorm -> fp32 transposed to [L,B,D] ----------------
__global__ __launch_bounds__(256) void lnf_kernel(const float* __restrict__ X,
                                                  const float* __restrict__ g,
                                                  const float* __restrict__ bta,
                                                  float* __restrict__ Out) {
  int w = threadIdx.x >> 5, lane = threadIdx.x & 31;
  int row = blockIdx.x * 8 + w;           // row = b*LL + l
  const float* x = X + (size_t)row * DD;
  float s = 0.f, ss = 0.f, vals[16];
#pragma unroll
  for (int j = 0; j < 16; ++j) { float v = x[lane + j * 32]; vals[j] = v; s += v; ss += v * v; }
#pragma unroll
  for (int o = 16; o >= 1; o >>= 1) { s += __shfl_xor(s, o); ss += __shfl_xor(ss, o); }
  float mean = s * (1.f / DD);
  float var = ss * (1.f / DD) - mean * mean;
  float rstd = rsqrtf(var + 1e-6f);
  int b = row >> 8, l = row & 255;
  float* o = Out + ((size_t)l * BB + b) * DD;
#pragma unroll
  for (int j = 0; j < 16; ++j) {
    int d = lane + j * 32;
    o[d] = (vals[j] - mean) * rstd * g[d] + bta[d];
  }
}

// ---------------- V transpose: [B,L,H,DH] -> [B,H,DH,L] ----------------
__global__ __launch_bounds__(256) void vtrans_kernel(const bf16* __restrict__ V,
                                                     bf16* __restrict__ Vt) {
  int i = blockIdx.x * 256 + threadIdx.x;      // i = ((b*LL+l)*HH+h)*DHD+dh
  int dh = i & 63;
  int t = i >> 6;
  int hh = t & 7; t >>= 3;
  int l = t & 255;
  int b = t >> 8;
  Vt[(((size_t)(b * HH + hh) * DHD) + dh) * LL + l] = V[i];
}

// ---------------- WMMA GEMM with async global->LDS staging ----------------
// C = A(MxK,bf16) * Bt^T(NxK,bf16) + bias
// RELU: apply relu; RESID: accumulate fp32 into Cres (residual), else write bf16 Cbf.
// Block: 256 thr = 8 waves; block tile 128x128; wave tile 64x32 (4x2 WMMA tiles).
// K staged through LDS in 32-wide double-buffered tiles via GLOBAL_LOAD_ASYNC_TO_LDS_B128
// (ASYNCcnt), rows padded to 40 bf16 (80B stride: 16B-aligned, conflict-free frag reads).
#define KSTEP 32
#define LDS_STRIDE 40
template <int RELU, int RESID>
__global__ __launch_bounds__(256) void gemm_kernel(const bf16* __restrict__ A,
                                                   const bf16* __restrict__ Bt,
                                                   const float* __restrict__ bias,
                                                   bf16* __restrict__ Cbf,
                                                   float* __restrict__ Cres,
                                                   int M, int N, int K) {
  __shared__ __align__(16) bf16 sA[2][128][LDS_STRIDE];
  __shared__ __align__(16) bf16 sB[2][128][LDS_STRIDE];

  const int lane = threadIdx.x & 31;
  const int w = threadIdx.x >> 5;
  const int wm = w & 1, wn = w >> 1;
  const int m0blk = blockIdx.y * 128;
  const int n0blk = blockIdx.x * 128;
  const int row = lane & 15, half = lane >> 4;

  // async-copy one 128x32 K-slice of A and of B into LDS buffer `buf`
  auto prefetch = [&](int buf, int kk) {
#pragma unroll
    for (int q = 0; q < 2; ++q) {
      int c = threadIdx.x * 2 + q;          // chunk id 0..511 (16B chunks)
      int r = c >> 2, col = (c & 3) * 8;    // 4 chunks per 32-elem row
      const bf16* ga = A + (size_t)(m0blk + r) * K + kk + col;
      unsigned la = (unsigned)(unsigned long long)&sA[buf][r][col];
      asm volatile("global_load_async_to_lds_b128 %0, %1, off"
                   :: "v"(la), "v"((unsigned long long)ga) : "memory");
      const bf16* gb = Bt + (size_t)(n0blk + r) * K + kk + col;
      unsigned lb = (unsigned)(unsigned long long)&sB[buf][r][col];
      asm volatile("global_load_async_to_lds_b128 %0, %1, off"
                   :: "v"(lb), "v"((unsigned long long)gb) : "memory");
    }
  };

  v8f acc[4][2];
#pragma unroll
  for (int i = 0; i < 4; ++i)
#pragma unroll
    for (int j = 0; j < 2; ++j)
#pragma unroll
      for (int r = 0; r < 8; ++r) acc[i][j][r] = 0.f;

  const int nsteps = K / KSTEP;
  prefetch(0, 0);
  for (int s = 0; s < nsteps; ++s) {
    const int buf = s & 1;
    if (s + 1 < nsteps) {
      prefetch(buf ^ 1, (s + 1) * KSTEP);
      asm volatile("s_wait_asynccnt 0x4" ::: "memory");  // tile s done, tile s+1 in flight
    } else {
      asm volatile("s_wait_asynccnt 0x0" ::: "memory");
    }
    __syncthreads();  // all waves' tile-s copies visible

    v16bf a[4], b[2];
#pragma unroll
    for (int i = 0; i < 4; ++i) {
      // A fragment (ISA 16-bit A 16x32): lane row, chunks at k=half*8 and half*8+16
      const bf16* p = &sA[buf][wm * 64 + 16 * i + row][half * 8];
      union { v16bf v; uv4 u[2]; } f;
      f.u[0] = *(const uv4*)p;
      f.u[1] = *(const uv4*)(p + 16);
      a[i] = f.v;
    }
#pragma unroll
    for (int j = 0; j < 2; ++j) {
      // B fragment from Bt (NxK): lane = column n, chunks at k=half*8 and half*8+16
      const bf16* p = &sB[buf][wn * 32 + 16 * j + row][half * 8];
      union { v16bf v; uv4 u[2]; } f;
      f.u[0] = *(const uv4*)p;
      f.u[1] = *(const uv4*)(p + 16);
      b[j] = f.v;
    }
#pragma unroll
    for (int i = 0; i < 4; ++i)
#pragma unroll
      for (int j = 0; j < 2; ++j)
        acc[i][j] = __builtin_amdgcn_wmma_f32_16x16x32_bf16(
            false, a[i], false, b[j], (short)0, acc[i][j], false, false);

    __syncthreads();  // all waves done reading buf before it is overwritten
  }

#pragma unroll
  for (int j = 0; j < 2; ++j) {
    int n = n0blk + wn * 32 + 16 * j + row;
    float bvv = bias[n];
#pragma unroll
    for (int i = 0; i < 4; ++i) {
#pragma unroll
      for (int r = 0; r < 8; ++r) {
        int m = m0blk + wm * 64 + 16 * i + half * 8 + r;
        float v = acc[i][j][r] + bvv;
        if (RELU) v = fmaxf(v, 0.f);
        if (RESID) Cres[(size_t)m * N + n] += v;
        else       Cbf[(size_t)m * N + n] = f2bf(v);
      }
    }
  }
}

// ---------------- WMMA attention: one wave per (b, h, 16-query tile) ----------------
__global__ __launch_bounds__(32) void attn_kernel(const bf16* __restrict__ Q,
                                                  const bf16* __restrict__ Kb,
                                                  const bf16* __restrict__ Vt,
                                                  const float* __restrict__ mask,
                                                  bf16* __restrict__ Ctx) {
  __shared__ __align__(16) float s_sc[16][LL];   // scores, 16 KB
  __shared__ __align__(16) bf16  s_p[16][LL];    // probs bf16, 8 KB
  const int lane = threadIdx.x & 31;
  const int row = lane & 15, half = lane >> 4;
  const int qt = blockIdx.x, hh = blockIdx.y, b = blockIdx.z;
  const int qbase = qt * 16;
  const float scale = 0.125f;  // 1/sqrt(DH=64)

  // Q tile: 16 x 64 -> two A fragments (K = dh)
  v16bf aq[2];
#pragma unroll
  for (int f = 0; f < 2; ++f) {
    const bf16* p = Q + ((size_t)(b * LL + qbase + row) * HH + hh) * DHD + f * 32 + half * 8;
    union { v16bf v; uv4 u[2]; } u;
    u.u[0] = *(const uv4*)p;
    u.u[1] = *(const uv4*)(p + 16);
    aq[f] = u.v;
  }

  // S = Q K^T * scale + mask, staged in LDS
  for (int kt = 0; kt < LL / 16; ++kt) {
    v16bf bk[2];
#pragma unroll
    for (int f = 0; f < 2; ++f) {
      const bf16* p = Kb + ((size_t)(b * LL + kt * 16 + row) * HH + hh) * DHD + f * 32 + half * 8;
      union { v16bf v; uv4 u[2]; } u;
      u.u[0] = *(const uv4*)p;
      u.u[1] = *(const uv4*)(p + 16);
      bk[f] = u.v;
    }
    v8f s;
#pragma unroll
    for (int r = 0; r < 8; ++r) s[r] = 0.f;
    s = __builtin_amdgcn_wmma_f32_16x16x32_bf16(false, aq[0], false, bk[0], (short)0, s, false, false);
    s = __builtin_amdgcn_wmma_f32_16x16x32_bf16(false, aq[1], false, bk[1], (short)0, s, false, false);
    float mv = mask[b * LL + kt * 16 + row];  // lane column = key index
#pragma unroll
    for (int r = 0; r < 8; ++r)
      s_sc[half * 8 + r][kt * 16 + row] = s[r] * scale + mv;
  }
  __syncthreads();

  // row softmax (lanes 0..15, one query row each) -> bf16 probs in LDS
  if (lane < 16) {
    float mx = -3.0e38f;
    for (int j = 0; j < LL; ++j) mx = fmaxf(mx, s_sc[lane][j]);
    float sum = 0.f;
    for (int j = 0; j < LL; ++j) { float e = __expf(s_sc[lane][j] - mx); s_sc[lane][j] = e; sum += e; }
    float inv = 1.f / sum;
    for (int j = 0; j < LL; ++j) s_p[lane][j] = f2bf(s_sc[lane][j] * inv);
  }
  __syncthreads();

  // ctx = P V : A = P (16 x 32 keys from LDS), B = Vt (dh x keys, row-major)
  v8f acc[4];
#pragma unroll
  for (int c = 0; c < 4; ++c)
#pragma unroll
    for (int r = 0; r < 8; ++r) acc[c][r] = 0.f;

  for (int kt = 0; kt < LL / 32; ++kt) {
    v16bf ap;
    {
      const bf16* p = &s_p[row][kt * 32 + half * 8];
      union { v16bf v; uv4 u[2]; } u;
      u.u[0] = *(const uv4*)p;
      u.u[1] = *(const uv4*)(p + 16);
      ap = u.v;
    }
#pragma unroll
    for (int c = 0; c < 4; ++c) {
      const bf16* p = Vt + (((size_t)(b * HH + hh) * DHD) + c * 16 + row) * LL + kt * 32 + half * 8;
      union { v16bf v; uv4 u[2]; } u;
      u.u[0] = *(const uv4*)p;
      u.u[1] = *(const uv4*)(p + 16);
      acc[c] = __builtin_amdgcn_wmma_f32_16x16x32_bf16(false, ap, false, u.v, (short)0, acc[c], false, false);
    }
  }
#pragma unroll
  for (int c = 0; c < 4; ++c) {
    int n = c * 16 + row;  // dh index
#pragma unroll
    for (int r = 0; r < 8; ++r) {
      int q = qbase + half * 8 + r;
      Ctx[((size_t)(b * LL + q) * HH + hh) * DHD + n] = f2bf(acc[c][r]);
    }
  }
}

// ---------------- host orchestration ----------------
extern "C" void kernel_launch(void* const* d_in, const int* in_sizes, int n_in,
                              void* d_out, int out_size, void* d_ws, size_t ws_size,
                              hipStream_t stream) {
  const int*   x      = (const int*)  d_in[0];
  const float* acoord = (const float*)d_in[1];
  const int*   atok   = (const int*)  d_in[2];
  const int*   aidx   = (const int*)  d_in[3];
  const int*   bidx   = (const int*)  d_in[4];
  const float* emb    = (const float*)d_in[5];
  const float* coordw = (const float*)d_in[6];
  const float* coordb = (const float*)d_in[7];
  const float* rate1  = (const float*)d_in[8];
  const float* rate2  = (const float*)d_in[9];
  const float* ln1g   = (const float*)d_in[10];
  const float* ln1b   = (const float*)d_in[11];
  const float* wq     = (const float*)d_in[12];
  const float* bq     = (const float*)d_in[13];
  const float* wk     = (const float*)d_in[14];
  const float* bk_    = (const float*)d_in[15];
  const float* wv     = (const float*)d_in[16];
  const float* bv     = (const float*)d_in[17];
  const float* wo     = (const float*)d_in[18];
  const float* bo     = (const float*)d_in[19];
  const float* ln2g   = (const float*)d_in[20];
  const float* ln2b   = (const float*)d_in[21];
  const float* w1     = (const float*)d_in[22];
  const float* b1     = (const float*)d_in[23];
  const float* w2     = (const float*)d_in[24];
  const float* b2     = (const float*)d_in[25];
  const float* lnfg   = (const float*)d_in[26];
  const float* lnfb   = (const float*)d_in[27];
  float* out = (float*)d_out;

  char* ws = (char*)d_ws;
  size_t off = 0;
  auto alloc = [&](size_t bytes) -> void* {
    void* p = ws + off;
    off += (bytes + 255) & ~(size_t)255;
    return p;
  };
  float* hbuf = (float*)alloc((size_t)MM * DD * 4);
  bf16* hn   = (bf16*)alloc((size_t)MM * DD * 2);
  bf16* qb   = (bf16*)alloc((size_t)MM * DD * 2);
  bf16* kbuf = (bf16*)alloc((size_t)MM * DD * 2);
  bf16* vbuf = (bf16*)alloc((size_t)MM * DD * 2);
  bf16* vt   = (bf16*)alloc((size_t)MM * DD * 2);
  bf16* ctx  = (bf16*)alloc((size_t)MM * DD * 2);
  bf16* ffb  = (bf16*)alloc((size_t)MM * FFD * 2);
  bf16* wqt  = (bf16*)alloc((size_t)NLAYER * DD * DD * 2);
  bf16* wkt  = (bf16*)alloc((size_t)NLAYER * DD * DD * 2);
  bf16* wvt  = (bf16*)alloc((size_t)NLAYER * DD * DD * 2);
  bf16* wot  = (bf16*)alloc((size_t)NLAYER * DD * DD * 2);
  bf16* w1t  = (bf16*)alloc((size_t)NLAYER * DD * FFD * 2);
  bf16* w2t  = (bf16*)alloc((size_t)NLAYER * FFD * DD * 2);
  float* maskb = (float*)alloc((size_t)MM * 4);

  // weights -> bf16, transposed to N x K (resident in 192 MB L2)
  int totD = NLAYER * DD * DD;
  wconv_kernel<<<totD / 256, 256, 0, stream>>>(wq, wqt, DD, DD, totD);
  wconv_kernel<<<totD / 256, 256, 0, stream>>>(wk, wkt, DD, DD, totD);
  wconv_kernel<<<totD / 256, 256, 0, stream>>>(wv, wvt, DD, DD, totD);
  wconv_kernel<<<totD / 256, 256, 0, stream>>>(wo, wot, DD, DD, totD);
  int totF = NLAYER * DD * FFD;
  wconv_kernel<<<totF / 256, 256, 0, stream>>>(w1, w1t, DD, FFD, totF);
  wconv_kernel<<<totF / 256, 256, 0, stream>>>(w2, w2t, FFD, DD, totF);

  embed_kernel<<<MM, 256, 0, stream>>>(x, emb, rate1, hbuf, maskb);
  atom_kernel<<<NTOT, 256, 0, stream>>>(acoord, atok, aidx, bidx, emb, coordw, coordb, rate2, hbuf);

  dim3 gD(DD / 128, MM / 128);
  dim3 gF(FFD / 128, MM / 128);
  for (int l = 0; l < NLAYER; ++l) {
    ln_kernel<<<MM / 8, 256, 0, stream>>>(hbuf, ln1g + l * DD, ln1b + l * DD, hn);
    gemm_kernel<0, 0><<<gD, 256, 0, stream>>>(hn, wqt + (size_t)l * DD * DD, bq + l * DD, qb,   nullptr, MM, DD, DD);
    gemm_kernel<0, 0><<<gD, 256, 0, stream>>>(hn, wkt + (size_t)l * DD * DD, bk_ + l * DD, kbuf, nullptr, MM, DD, DD);
    gemm_kernel<0, 0><<<gD, 256, 0, stream>>>(hn, wvt + (size_t)l * DD * DD, bv + l * DD, vbuf, nullptr, MM, DD, DD);
    vtrans_kernel<<<(MM * DD) / 256, 256, 0, stream>>>(vbuf, vt);
    attn_kernel<<<dim3(LL / 16, HH, BB), 32, 0, stream>>>(qb, kbuf, vt, maskb, ctx);
    gemm_kernel<0, 1><<<gD, 256, 0, stream>>>(ctx, wot + (size_t)l * DD * DD, bo + l * DD, nullptr, hbuf, MM, DD, DD);
    ln_kernel<<<MM / 8, 256, 0, stream>>>(hbuf, ln2g + l * DD, ln2b + l * DD, hn);
    gemm_kernel<1, 0><<<gF, 256, 0, stream>>>(hn, w1t + (size_t)l * DD * FFD, b1 + l * FFD, ffb, nullptr, MM, FFD, DD);
    gemm_kernel<0, 1><<<gD, 256, 0, stream>>>(ffb, w2t + (size_t)l * FFD * DD, b2 + l * DD, nullptr, hbuf, MM, DD, FFD);
  }
  lnf_kernel<<<MM / 8, 256, 0, stream>>>(hbuf, lnfg, lnfb, out);
}
